// Head_2010044694565
// MI455X (gfx1250) — matrix operand order repaired and checked by reading, not compiled, not run on hardware
//
#include <hip/hip_runtime.h>
#include <math.h>

// Problem constants (match reference)
#define BB 4
#define TT 4096
#define CC 512
#define HH 64

typedef __attribute__((ext_vector_type(16))) __bf16 v16bf;
typedef __attribute__((ext_vector_type(8)))  __bf16 v8bf;
typedef __attribute__((ext_vector_type(8)))  float  v8f;

#define NEG_INF (-__builtin_inff())

__device__ __forceinline__ v8f wmma_bf16(v16bf a, v16bf b, v8f c) {
  // D = A(16x32 bf16) * B(32x16 bf16) + C(16x16 f32)
  return __builtin_amdgcn_wmma_f32_16x16x32_bf16(
      /*neg_a=*/false, a, /*neg_b=*/false, b,
      /*c_mod=*/(short)0, c, /*reuse_a=*/false, /*reuse_b=*/false);
}

__device__ __forceinline__ v8f v8f_zero() {
  v8f z = {0.f, 0.f, 0.f, 0.f, 0.f, 0.f, 0.f, 0.f};
  return z;
}

// ---------------------------------------------------------------------------
// Kernel 0: W[C,H] f32  ->  Wt[H,C] bf16 (so WMMA B-frags are contiguous)
// ---------------------------------------------------------------------------
__global__ void wt_kernel(const float* __restrict__ Wk,
                          const float* __restrict__ Wq,
                          const float* __restrict__ Wv,
                          __bf16* __restrict__ WtQ,
                          __bf16* __restrict__ WtK,
                          __bf16* __restrict__ WtV) {
  int idx = blockIdx.x * blockDim.x + threadIdx.x;
  if (idx >= CC * HH) return;
  int c = idx / HH;
  int h = idx % HH;
  WtQ[h * CC + c] = (__bf16)Wq[idx];
  WtK[h * CC + c] = (__bf16)Wk[idx];
  WtV[h * CC + c] = (__bf16)Wv[idx];
}

// ---------------------------------------------------------------------------
// Kernel 1: QKV projection. One wave per 16 rows of (b,t).
//   q,k stored row-major [B,T,64] bf16; v stored transposed [B,64,T] bf16.
// A-frag (16-bit 16x32): lane row m = lane%16, g = lane/16,
//   elements e=0..15 -> K = 16g+e (contiguous, same order as B)
// B-frag: lane col n = lane%16, element e -> K = 16g+e (contiguous in Wt row)
// C/D: lane col = lane%16, VGPR v -> row v + 8g
// Fully unrolled K loop; every fragment is a fresh local with constant
// indices only (register-resident), and each step gathers its 12 B-frags
// before issuing the 12 WMMAs so load clauses run ahead of the matrix ops.
// ---------------------------------------------------------------------------
__global__ void __launch_bounds__(256)
qkv_kernel(const float* __restrict__ x,
           const __bf16* __restrict__ WtQ,
           const __bf16* __restrict__ WtK,
           const __bf16* __restrict__ WtV,
           __bf16* __restrict__ qws,
           __bf16* __restrict__ kws,
           __bf16* __restrict__ vtws) {
  const int lane = threadIdx.x & 31;
  const int wv   = threadIdx.x >> 5;
  const int g    = lane >> 4;
  const int m    = lane & 15;
  const int tile = blockIdx.x * 8 + wv;        // 0 .. B*T/16 - 1
  const int b    = tile / (TT / 16);
  const int t0   = (tile % (TT / 16)) * 16;

  v8f accQ[4], accK[4], accV[4];
#pragma unroll
  for (int nt = 0; nt < 4; ++nt) {
    accQ[nt] = v8f_zero();
    accK[nt] = v8f_zero();
    accV[nt] = v8f_zero();
  }

  const float* xrow = x + ((size_t)b * TT + t0 + m) * CC;

#pragma unroll
  for (int c0 = 0; c0 < CC; c0 += 32) {
    // A-fragment: x tile rows, converted f32 -> bf16 in registers
    union { v16bf v; __bf16 e[16]; } a;
#pragma unroll
    for (int j = 0; j < 8; ++j) {
      a.e[j]     = (__bf16)xrow[c0 + 16 * g + j];
      a.e[8 + j] = (__bf16)xrow[c0 + 16 * g + 8 + j];
    }
    // gather all 12 B-frags first so the load clause issues ahead of the MACs
    v16bf bq[4], bk[4], bv[4];
#pragma unroll
    for (int nt = 0; nt < 4; ++nt) {
      size_t woff = (size_t)(nt * 16 + m) * CC + c0 + 16 * g;
      bq[nt] = *(const v16bf*)(WtQ + woff);
      bk[nt] = *(const v16bf*)(WtK + woff);
      bv[nt] = *(const v16bf*)(WtV + woff);
    }
#pragma unroll
    for (int nt = 0; nt < 4; ++nt) {
      accQ[nt] = wmma_bf16(a.v, bq[nt], accQ[nt]);
      accK[nt] = wmma_bf16(a.v, bk[nt], accK[nt]);
      accV[nt] = wmma_bf16(a.v, bv[nt], accV[nt]);
    }
  }

#pragma unroll
  for (int nt = 0; nt < 4; ++nt) {
#pragma unroll
    for (int v = 0; v < 8; ++v) {
      size_t off = ((size_t)b * TT + t0 + 8 * g + v) * HH + nt * 16 + m;
      qws[off] = (__bf16)accQ[nt][v];
      kws[off] = (__bf16)accK[nt][v];
    }
    // v transposed: Vt[b][h][t]; rows v=0..7 are contiguous in t -> b128 store
    v8bf pv;
#pragma unroll
    for (int v = 0; v < 8; ++v) pv[v] = (__bf16)accV[nt][v];
    *(v8bf*)(vtws + ((size_t)b * HH + nt * 16 + m) * TT + t0 + 8 * g) = pv;
  }
}

// ---------------------------------------------------------------------------
// Kernel 2: causal flash attention, split-K across 4 waves per query tile.
// Wave w handles key blocks w, w+4, w+8, ... (balanced causal triangle).
// Per 32-key block: S = Q K^T (4 WMMA), online softmax, O += P V (4 WMMA).
// P transposed (D-layout -> A-layout) through per-wave LDS staging.
// Partial (O, m, l) merged across waves via LDS log-sum-exp combine.
// ---------------------------------------------------------------------------
__global__ void __launch_bounds__(128)
attn_kernel(const __bf16* __restrict__ qws,
            const __bf16* __restrict__ kws,
            const __bf16* __restrict__ vtws,
            float* __restrict__ out) {
  __shared__ __align__(16) __bf16 Pl[4][16][32];   // per-wave P staging (4 KB)
  __shared__ __align__(16) float  Of[4][16][64];   // partial O        (16 KB)
  __shared__ float Ml[4][16], Ll[4][16];           // partial m, l

  const int lane = threadIdx.x & 31;
  const int wv   = threadIdx.x >> 5;
  const int g    = lane >> 4;
  const int m    = lane & 15;
  const int tile = blockIdx.x;                 // 0 .. B*T/16 - 1
  const int b    = tile / (TT / 16);
  const int t0   = (tile % (TT / 16)) * 16;

  union U16 { v16bf v; v8bf h[2]; };

  // Q A-frags for the two 32-wide hs chunks (HS = 64)
  const __bf16* qrow = qws + ((size_t)b * TT + t0 + m) * HH;
  U16 qa[2];
#pragma unroll
  for (int hc = 0; hc < 2; ++hc) {
    qa[hc].h[0] = *(const v8bf*)(qrow + hc * 32 + 16 * g);
    qa[hc].h[1] = *(const v8bf*)(qrow + hc * 32 + 16 * g + 8);
  }

  v8f o[4];
#pragma unroll
  for (int nt = 0; nt < 4; ++nt) o[nt] = v8f_zero();
  float mrow[8], lrow[8];
#pragma unroll
  for (int v = 0; v < 8; ++v) { mrow[v] = NEG_INF; lrow[v] = 0.f; }

  const float scale = 0.125f;                  // 1/sqrt(64)
  const int   nblk  = t0 / 32 + 1;             // causal: keys <= t0+15

  const __bf16* kbase  = kws + (size_t)b * TT * HH;
  const __bf16* vtbase = vtws + (size_t)b * HH * TT;

  for (int blk = wv; blk < nblk; blk += 4) {
    const int s0 = blk * 32;

    // prefetch this wave's next key block (lowers to global_prefetch_b8)
    if (blk + 4 < nblk) {
      const int sn = s0 + 128;
      __builtin_prefetch(kbase + (size_t)(sn + lane) * HH, 0, 3);       // K rows
      __builtin_prefetch(vtbase + (size_t)(2 * lane) * TT + sn, 0, 3);  // Vt rows
      __builtin_prefetch(vtbase + (size_t)(2 * lane + 1) * TT + sn, 0, 3);
    }

    // ---- S = Q K^T for 32 keys (two 16x16 tiles) ----
    v8f s[2];
#pragma unroll
    for (int kt = 0; kt < 2; ++kt) {
      const __bf16* krow = kbase + (size_t)(s0 + kt * 16 + m) * HH;
      v16bf kb0 = *(const v16bf*)(krow + 16 * g);        // hs 0..31 chunk
      v16bf kb1 = *(const v16bf*)(krow + 32 + 16 * g);   // hs 32..63 chunk
      s[kt] = wmma_bf16(qa[0].v, kb0, v8f_zero());
      s[kt] = wmma_bf16(qa[1].v, kb1, s[kt]);
    }

    // ---- scale + causal mask (diagonal block only) ----
    const bool diag = (s0 + 31 >= t0);
    float rmax[8];
#pragma unroll
    for (int v = 0; v < 8; ++v) {
      const int row = t0 + 8 * g + v;
#pragma unroll
      for (int kt = 0; kt < 2; ++kt) {
        const int col = s0 + kt * 16 + m;
        float val = s[kt][v] * scale;
        val = (diag && (col > row)) ? NEG_INF : val;
        s[kt][v] = val;
      }
      rmax[v] = fmaxf(s[0][v], s[1][v]);
    }
    // row-wise max across the 16 lanes holding this row's columns
#pragma unroll
    for (int off = 1; off < 16; off <<= 1)
#pragma unroll
      for (int v = 0; v < 8; ++v)
        rmax[v] = fmaxf(rmax[v], __shfl_xor(rmax[v], off, 32));

    // ---- online softmax update; write P (bf16) to LDS in row-major ----
    float alpha[8], psum[8];
#pragma unroll
    for (int v = 0; v < 8; ++v) {
      const float mnew = fmaxf(mrow[v], rmax[v]);
      alpha[v] = __expf(mrow[v] - mnew);
      mrow[v]  = mnew;
      const float p0 = __expf(s[0][v] - mnew);
      const float p1 = __expf(s[1][v] - mnew);
      Pl[wv][8 * g + v][m]      = (__bf16)p0;
      Pl[wv][8 * g + v][16 + m] = (__bf16)p1;
      psum[v] = p0 + p1;
    }
#pragma unroll
    for (int off = 1; off < 16; off <<= 1)
#pragma unroll
      for (int v = 0; v < 8; ++v)
        psum[v] += __shfl_xor(psum[v], off, 32);
#pragma unroll
    for (int v = 0; v < 8; ++v)
      lrow[v] = lrow[v] * alpha[v] + psum[v];
#pragma unroll
    for (int nt = 0; nt < 4; ++nt)
#pragma unroll
      for (int v = 0; v < 8; ++v)
        o[nt][v] *= alpha[v];

    // ---- read P back in A-layout (LDS is in-order within a wave) ----
    __builtin_amdgcn_wave_barrier();
    U16 pa;
    pa.h[0] = *(const v8bf*)&Pl[wv][m][16 * g];
    pa.h[1] = *(const v8bf*)&Pl[wv][m][16 * g + 8];
    __builtin_amdgcn_wave_barrier();

    // ---- O += P V  (V B-frags contiguous thanks to transposed storage) ----
#pragma unroll
    for (int nt = 0; nt < 4; ++nt) {
      v16bf vb = *(const v16bf*)(vtbase + (size_t)(nt * 16 + m) * TT +
                                 s0 + 16 * g);
      o[nt] = wmma_bf16(pa.v, vb, o[nt]);
    }
  }

  // ---- publish this wave's partial (O, m, l) ----
#pragma unroll
  for (int nt = 0; nt < 4; ++nt)
#pragma unroll
    for (int v = 0; v < 8; ++v)
      Of[wv][8 * g + v][nt * 16 + m] = o[nt][v];
  if (m == 0) {
#pragma unroll
    for (int v = 0; v < 8; ++v) {
      Ml[wv][8 * g + v] = mrow[v];
      Ll[wv][8 * g + v] = lrow[v];
    }
  }
  __syncthreads();

  // ---- merge the 4 partials (log-sum-exp combine), store f32 ----
  for (int e = threadIdx.x; e < 16 * HH; e += 128) {
    const int row = e >> 6;        // 0..15
    const int col = e & 63;        // 0..63
    float M = Ml[0][row];
#pragma unroll
    for (int w = 1; w < 4; ++w) M = fmaxf(M, Ml[w][row]);
    float L = 0.f, O = 0.f;
#pragma unroll
    for (int w = 0; w < 4; ++w) {
      const float be = __expf(Ml[w][row] - M);   // 0 for idle waves (m=-inf)
      L += be * Ll[w][row];
      O += be * Of[w][row][col];
    }
    out[((size_t)b * TT + t0 + row) * HH + col] = O / L;
  }
}

// ---------------------------------------------------------------------------
// Launch. Workspace layout (bytes):
//   [0, 2MB)   q   bf16 [B,T,64]
//   [2MB,4MB)  k   bf16 [B,T,64]
//   [4MB,6MB)  vT  bf16 [B,64,T]
//   [6MB, +192KB) WtQ/WtK/WtV bf16 [64,512] each
// ---------------------------------------------------------------------------
extern "C" void kernel_launch(void* const* d_in, const int* in_sizes, int n_in,
                              void* d_out, int out_size, void* d_ws,
                              size_t ws_size, hipStream_t stream) {
  const float* x  = (const float*)d_in[0];
  const float* Wk = (const float*)d_in[1];
  const float* Wq = (const float*)d_in[2];
  const float* Wv = (const float*)d_in[3];
  float* out = (float*)d_out;

  char* ws = (char*)d_ws;
  const size_t qkv_bytes = (size_t)BB * TT * HH * sizeof(__bf16); // 2 MB
  __bf16* qws  = (__bf16*)(ws);
  __bf16* kws  = (__bf16*)(ws + qkv_bytes);
  __bf16* vtws = (__bf16*)(ws + 2 * qkv_bytes);
  __bf16* WtQ  = (__bf16*)(ws + 3 * qkv_bytes);
  __bf16* WtK  = WtQ + (size_t)CC * HH;
  __bf16* WtV  = WtK + (size_t)CC * HH;

  wt_kernel<<<(CC * HH + 255) / 256, 256, 0, stream>>>(Wk, Wq, Wv, WtQ, WtK,
                                                       WtV);
  qkv_kernel<<<(BB * TT / 16) / 8, 256, 0, stream>>>(x, WtQ, WtK, WtV, qws,
                                                     kws, vtws);
  attn_kernel<<<BB * TT / 16, 128, 0, stream>>>(qws, kws, vtws, out);
}
